// Attention_26688926778055
// MI455X (gfx1250) — matrix operand order repaired and checked by reading, not compiled, not run on hardware
//
#include <hip/hip_runtime.h>
#include <hip/hip_bf16.h>

// ---------------------------------------------------------------------------
// Attention (B=2, S=2048, H=2048, NH=16, HD=128) for gfx1250 (MI455X, wave32)
// f32 -> bf16 operands, v_wmma_f32_16x16x32_bf16, f32 accumulate.
// GEMM: 128x128 macro-tile, LDS double-buffered via global_load_async_to_lds.
// Epilogue selected at compile time (f32 store / bf16 store / fused RoPE).
// ---------------------------------------------------------------------------

typedef __bf16 bf16_t;
typedef __attribute__((ext_vector_type(16))) bf16_t v16bf;
typedef __attribute__((ext_vector_type(4)))  bf16_t v4bf;
typedef __attribute__((ext_vector_type(8)))  float  v8f;
typedef __attribute__((ext_vector_type(4)))  float  v4f;
typedef __attribute__((ext_vector_type(4)))  unsigned int u32x4;

#define B_   2
#define S_   2048
#define H_   2048
#define NH_  16
#define HD_  128
#define MROWS (B_ * S_)          // 4096

#define ATTN_SCALE    0.08838834764831845f   // 1/sqrt(128)
#define LOG_BASE_O_HD 0.0719557841f          // ln(10000)/128

#define EPI_F32  0
#define EPI_BF16 1
#define EPI_ROPE 2

static __device__ __forceinline__ v8f wmma_bf16(v16bf a, v16bf b, v8f c) {
  return __builtin_amdgcn_wmma_f32_16x16x32_bf16(
      false, a, false, b, (short)0, c, false, false);
}

static __device__ __forceinline__ v8f v8f_zero() {
  v8f v = {0.f, 0.f, 0.f, 0.f, 0.f, 0.f, 0.f, 0.f};
  return v;
}

// one per-lane 16B async global->LDS copy (ASYNCcnt-tracked)
static __device__ __forceinline__ void async_copy_b128(unsigned lds_off,
                                                       int gvoff_bytes,
                                                       unsigned long long gbase) {
  asm volatile("global_load_async_to_lds_b128 %0, %1, %2"
               :: "v"(lds_off), "v"(gvoff_bytes), "s"(gbase) : "memory");
}
static __device__ __forceinline__ void wait_async0() {
  asm volatile("s_wait_asynccnt 0x0" ::: "memory");
}

// ---------------------------------------------------------------------------
// f32 -> bf16 convert (x4 vectorized; all sizes divisible by 4)
// ---------------------------------------------------------------------------
__global__ void cvt_f32_bf16_kernel(const v4f* __restrict__ src,
                                    v4bf* __restrict__ dst, int n4) {
  int i = blockIdx.x * blockDim.x + threadIdx.x;
  if (i < n4) {
    v4f x = src[i];
    v4bf y;
    y[0] = (bf16_t)x[0]; y[1] = (bf16_t)x[1];
    y[2] = (bf16_t)x[2]; y[3] = (bf16_t)x[3];
    dst[i] = y;
  }
}

// ---------------------------------------------------------------------------
// V transpose: Vb[b,s,h*HD+d] -> Vt[(b*NH+h)*HD+d, s]   (bf16)
// ---------------------------------------------------------------------------
__global__ void transpose_v_kernel(const bf16_t* __restrict__ Vb,
                                   bf16_t* __restrict__ Vt) {
  size_t idx = (size_t)blockIdx.x * blockDim.x + threadIdx.x; // over B*NH*HD*S
  int    s  = (int)(idx & (S_ - 1));
  size_t r  = idx >> 11;            // / S_
  int    d  = (int)(r & (HD_ - 1));
  size_t bh = r >> 7;               // / HD_
  int    h  = (int)(bh & (NH_ - 1));
  int    b  = (int)(bh >> 4);
  Vt[idx] = Vb[(size_t)(b * S_ + s) * H_ + h * HD_ + d];
}

// ---------------------------------------------------------------------------
// NT GEMM: C[M,N] = A[M,K] * W[N,K]^T   (bf16 operands, f32 accumulate)
// Block 256 thr (8 waves), macro tile 128M x 128N, BK=32, LDS double buffer
// filled with global_load_async_to_lds_b128. Wave tile 32x64 (2x4 WMMA accs).
// LDS rows padded to 40 bf16 -> conflict-free ds_load_b128 fragment reads.
// EPI is a compile-time parameter -> straight-line epilogue stores.
// ---------------------------------------------------------------------------
#define BK_ 32
#define LDP 40   // padded row length (bf16 elements)

template <int EPI>
__global__ __launch_bounds__(256)
void gemm_nt_kernel(const bf16_t* __restrict__ A, const bf16_t* __restrict__ W,
                    void* __restrict__ Cout, int M, int N, int K) {
  __shared__ __align__(16) bf16_t Asm[2][128][LDP];
  __shared__ __align__(16) bf16_t Bsm[2][128][LDP];

  const int tid  = threadIdx.x;
  const int lane = tid & 31;
  const int wave = tid >> 5;
  const int l15  = lane & 15;
  const int half = lane >> 4;
  const int wm = wave & 3;          // 4 waves along M
  const int wn = wave >> 2;         // 2 waves along N
  const int mBlk = blockIdx.y * 128;
  const int nBlk = blockIdx.x * 128;

  // async stage: 128x32 bf16 tile = 512 x 16B chunks, 2 chunks/thread/matrix
  const int row0 = (tid + 0)   >> 2, seg0 = (tid + 0)   & 3;
  const int row1 = (tid + 256) >> 2, seg1 = (tid + 256) & 3;

  auto stage = [&](int k0, int buf) {
    unsigned long long abase =
        (unsigned long long)(A + (size_t)mBlk * K + k0);
    unsigned long long bbase =
        (unsigned long long)(W + (size_t)nBlk * K + k0);
    int v0 = (row0 * K + seg0 * 8) * 2;
    int v1 = (row1 * K + seg1 * 8) * 2;
    async_copy_b128((unsigned)(uintptr_t)&Asm[buf][row0][seg0 * 8], v0, abase);
    async_copy_b128((unsigned)(uintptr_t)&Asm[buf][row1][seg1 * 8], v1, abase);
    async_copy_b128((unsigned)(uintptr_t)&Bsm[buf][row0][seg0 * 8], v0, bbase);
    async_copy_b128((unsigned)(uintptr_t)&Bsm[buf][row1][seg1 * 8], v1, bbase);
  };

  v8f acc[2][4];
  #pragma unroll
  for (int i = 0; i < 2; ++i)
    #pragma unroll
    for (int j = 0; j < 4; ++j) acc[i][j] = v8f_zero();

  stage(0, 0);
  wait_async0();
  __syncthreads();

  int buf = 0;
  for (int k0 = 0; k0 < K; k0 += BK_) {
    const bool more = (k0 + BK_) < K;
    if (more) stage(k0 + BK_, buf ^ 1);

    v16bf af[2], bf[4];
    #pragma unroll
    for (int i = 0; i < 2; ++i) {
      const bf16_t* ar = &Asm[buf][wm * 32 + i * 16 + l15][0];
      ((u32x4*)&af[i])[0] = *(const u32x4*)(ar + half * 8);
      ((u32x4*)&af[i])[1] = *(const u32x4*)(ar + 16 + half * 8);
    }
    #pragma unroll
    for (int j = 0; j < 4; ++j) {
      const bf16_t* br = &Bsm[buf][wn * 64 + j * 16 + l15][0];
      ((u32x4*)&bf[j])[0] = *(const u32x4*)(br + half * 8);
      ((u32x4*)&bf[j])[1] = *(const u32x4*)(br + 16 + half * 8);
    }
    #pragma unroll
    for (int i = 0; i < 2; ++i)
      #pragma unroll
      for (int j = 0; j < 4; ++j) acc[i][j] = wmma_bf16(af[i], bf[j], acc[i][j]);

    if (more) wait_async0();
    __syncthreads();
    buf ^= 1;
  }

  // C-fragment: lane holds col ..+l15, rows ..+8*half+e
  #pragma unroll
  for (int i = 0; i < 2; ++i) {
    #pragma unroll
    for (int j = 0; j < 4; ++j) {
      const int mbase = mBlk + wm * 32 + i * 16 + half * 8;
      const int col   = nBlk + wn * 64 + j * 16 + l15;
      if (EPI == EPI_F32) {
        float* outp = (float*)Cout + (size_t)mbase * N + col;
        #pragma unroll
        for (int e = 0; e < 8; ++e) outp[(size_t)e * N] = acc[i][j][e];
      } else if (EPI == EPI_BF16) {
        bf16_t* outp = (bf16_t*)Cout + (size_t)mbase * N + col;
        #pragma unroll
        for (int e = 0; e < 8; ++e) outp[(size_t)e * N] = (bf16_t)acc[i][j][e];
      } else { // EPI_ROPE: rotary pair lives in adjacent lane
        const int d = col & (HD_ - 1);
        const float invf = __expf(-(float)(d & ~1) * LOG_BASE_O_HD); // per-col
        bf16_t* outp = (bf16_t*)Cout + (size_t)mbase * N + col;
        #pragma unroll
        for (int e = 0; e < 8; ++e) {
          const float v = acc[i][j][e];
          const float partner = __shfl_xor(v, 1);
          const float ang = (float)((mbase + e) & (S_ - 1)) * invf;
          const float cc = __cosf(ang), ss = __sinf(ang);
          const float r =
              (d & 1) ? (partner * ss + v * cc) : (v * cc - partner * ss);
          outp[(size_t)e * N] = (bf16_t)r;
        }
      }
    }
  }
}

// ---------------------------------------------------------------------------
// Flash attention: grid (S/64, B*NH), block 128 (4 waves), wave = 16 queries.
// Online softmax with 16-lane shuffle reductions; P re-laid out via per-wave
// LDS tile (C-frag -> A-frag); causal loop bound per wave. K/V are L2-resident.
// ---------------------------------------------------------------------------
__global__ __launch_bounds__(128)
void flash_attn_kernel(const bf16_t* __restrict__ Q, const bf16_t* __restrict__ Kc,
                       const bf16_t* __restrict__ Vt, bf16_t* __restrict__ O) {
  __shared__ __align__(16) bf16_t plds[4][16][32];

  const int lane = threadIdx.x & 31;
  const int wave = threadIdx.x >> 5;
  const int l15  = lane & 15;
  const int half = lane >> 4;
  const int bh = blockIdx.y;
  const int b = bh / NH_, h = bh % NH_;
  const int q0 = blockIdx.x * 64 + wave * 16;

  v16bf aq[4];
  {
    const bf16_t* qp = Q + (size_t)(b * S_ + q0 + l15) * H_ + h * HD_ + half * 8;
    #pragma unroll
    for (int c = 0; c < 4; ++c) {
      ((u32x4*)&aq[c])[0] = *(const u32x4*)(qp + 32 * c);
      ((u32x4*)&aq[c])[1] = *(const u32x4*)(qp + 32 * c + 16);
    }
  }

  v8f acc[8];
  float mrow[8], lrow[8], pv0[8], pv1[8], alph[8];
  #pragma unroll
  for (int t = 0; t < 8; ++t) acc[t] = v8f_zero();
  #pragma unroll
  for (int e = 0; e < 8; ++e) { mrow[e] = -1e30f; lrow[e] = 0.f; }

  const bf16_t* kbase = Kc + (size_t)(b * S_) * H_ + h * HD_;
  const bf16_t* vtb   = Vt + (size_t)bh * HD_ * S_;

  for (int kb = 0; kb < q0 + 16; kb += 32) {
    v8f s0 = v8f_zero(), s1 = v8f_zero();
    #pragma unroll
    for (int c = 0; c < 4; ++c) {
      v16bf bk0, bk1;
      const bf16_t* kp0 = kbase + (size_t)(kb + l15) * H_ + half * 8 + 32 * c;
      const bf16_t* kp1 = kp0 + (size_t)16 * H_;
      ((u32x4*)&bk0)[0] = *(const u32x4*)(kp0);
      ((u32x4*)&bk0)[1] = *(const u32x4*)(kp0 + 16);
      ((u32x4*)&bk1)[0] = *(const u32x4*)(kp1);
      ((u32x4*)&bk1)[1] = *(const u32x4*)(kp1 + 16);
      s0 = wmma_bf16(aq[c], bk0, s0);
      s1 = wmma_bf16(aq[c], bk1, s1);
    }

    #pragma unroll
    for (int e = 0; e < 8; ++e) {
      const int row = q0 + half * 8 + e;
      float v0 = s0[e] * ATTN_SCALE;
      float v1 = s1[e] * ATTN_SCALE;
      if (kb + l15 > row)      v0 = -10000.0f;
      if (kb + 16 + l15 > row) v1 = -10000.0f;
      float mx = fmaxf(v0, v1);
      #pragma unroll
      for (int off = 1; off < 16; off <<= 1) mx = fmaxf(mx, __shfl_xor(mx, off));
      const float mnew  = fmaxf(mrow[e], mx);
      const float alpha = __expf(mrow[e] - mnew);
      const float p0 = __expf(v0 - mnew);
      const float p1 = __expf(v1 - mnew);
      float rs = p0 + p1;
      #pragma unroll
      for (int off = 1; off < 16; off <<= 1) rs += __shfl_xor(rs, off);
      lrow[e] = lrow[e] * alpha + rs;
      mrow[e] = mnew;
      alph[e] = alpha;
      pv0[e] = p0; pv1[e] = p1;
    }
    #pragma unroll
    for (int t = 0; t < 8; ++t)
      #pragma unroll
      for (int e = 0; e < 8; ++e) acc[t][e] *= alph[e];

    #pragma unroll
    for (int e = 0; e < 8; ++e) {
      const int r = half * 8 + e;
      plds[wave][r][l15]      = (bf16_t)pv0[e];
      plds[wave][r][16 + l15] = (bf16_t)pv1[e];
    }
    v16bf ap;
    ((u32x4*)&ap)[0] = *(const u32x4*)&plds[wave][l15][half * 8];
    ((u32x4*)&ap)[1] = *(const u32x4*)&plds[wave][l15][16 + half * 8];

    #pragma unroll
    for (int t = 0; t < 8; ++t) {
      const bf16_t* vp = vtb + (size_t)(t * 16 + l15) * S_ + kb + half * 8;
      v16bf bv;
      ((u32x4*)&bv)[0] = *(const u32x4*)(vp);
      ((u32x4*)&bv)[1] = *(const u32x4*)(vp + 16);
      acc[t] = wmma_bf16(ap, bv, acc[t]);
    }
  }

  #pragma unroll
  for (int t = 0; t < 8; ++t) {
    #pragma unroll
    for (int e = 0; e < 8; ++e) {
      const int row = q0 + half * 8 + e;
      const float val = acc[t][e] / lrow[e];
      O[(size_t)(b * S_ + row) * H_ + h * HD_ + t * 16 + l15] = (bf16_t)val;
    }
  }
}

// ---------------------------------------------------------------------------
// Host orchestration
// ---------------------------------------------------------------------------
extern "C" void kernel_launch(void* const* d_in, const int* in_sizes, int n_in,
                              void* d_out, int out_size, void* d_ws, size_t ws_size,
                              hipStream_t stream) {
  (void)in_sizes; (void)n_in; (void)out_size; (void)ws_size;
  const float* hs = (const float*)d_in[0];
  const float* Wq = (const float*)d_in[1];
  const float* Wk = (const float*)d_in[2];
  const float* Wv = (const float*)d_in[3];
  const float* Wd = (const float*)d_in[4];
  float* out = (float*)d_out;

  char* ws = (char*)d_ws;
  size_t off = 0;
  auto alloc = [&](size_t bytes) -> void* {
    void* p = ws + off;
    off += (bytes + 255) & ~(size_t)255;
    return p;
  };
  bf16_t* Xb  = (bf16_t*)alloc((size_t)MROWS * H_ * 2);
  bf16_t* Wqb = (bf16_t*)alloc((size_t)H_ * H_ * 2);
  bf16_t* Wkb = (bf16_t*)alloc((size_t)H_ * H_ * 2);
  bf16_t* Wvb = (bf16_t*)alloc((size_t)H_ * H_ * 2);
  bf16_t* Wdb = (bf16_t*)alloc((size_t)H_ * H_ * 2);
  bf16_t* Qb  = (bf16_t*)alloc((size_t)MROWS * H_ * 2);
  bf16_t* Kb  = (bf16_t*)alloc((size_t)MROWS * H_ * 2);
  bf16_t* Vb  = (bf16_t*)alloc((size_t)MROWS * H_ * 2);
  bf16_t* Vt  = (bf16_t*)alloc((size_t)MROWS * H_ * 2);
  bf16_t* Cb  = (bf16_t*)alloc((size_t)MROWS * H_ * 2);

  const int nHS4 = MROWS * H_ / 4;
  const int nW4  = H_ * H_ / 4;
  cvt_f32_bf16_kernel<<<(nHS4 + 255) / 256, 256, 0, stream>>>((const v4f*)hs, (v4bf*)Xb, nHS4);
  cvt_f32_bf16_kernel<<<(nW4 + 255) / 256, 256, 0, stream>>>((const v4f*)Wq, (v4bf*)Wqb, nW4);
  cvt_f32_bf16_kernel<<<(nW4 + 255) / 256, 256, 0, stream>>>((const v4f*)Wk, (v4bf*)Wkb, nW4);
  cvt_f32_bf16_kernel<<<(nW4 + 255) / 256, 256, 0, stream>>>((const v4f*)Wv, (v4bf*)Wvb, nW4);
  cvt_f32_bf16_kernel<<<(nW4 + 255) / 256, 256, 0, stream>>>((const v4f*)Wd, (v4bf*)Wdb, nW4);

  dim3 gg(H_ / 128, MROWS / 128);   // (16, 32)
  gemm_nt_kernel<EPI_ROPE><<<gg, 256, 0, stream>>>(Xb, Wqb, (void*)Qb, MROWS, H_, H_);
  gemm_nt_kernel<EPI_ROPE><<<gg, 256, 0, stream>>>(Xb, Wkb, (void*)Kb, MROWS, H_, H_);
  gemm_nt_kernel<EPI_BF16><<<gg, 256, 0, stream>>>(Xb, Wvb, (void*)Vb, MROWS, H_, H_);

  transpose_v_kernel<<<(MROWS * H_) / 256, 256, 0, stream>>>(Vb, Vt);

  flash_attn_kernel<<<dim3(S_ / 64, B_ * NH_), 128, 0, stream>>>(Qb, Kb, Vt, Cb);

  gemm_nt_kernel<EPI_F32><<<gg, 256, 0, stream>>>(Cb, Wdb, (void*)out, MROWS, H_, H_);
}
